// MemoryModule_64922725646514
// MI455X (gfx1250) — matrix operand order repaired and checked by reading, not compiled, not run on hardware
//
#include <hip/hip_runtime.h>

typedef __attribute__((ext_vector_type(16))) _Float16 v16h;
typedef __attribute__((ext_vector_type(2)))  _Float16 h2;
typedef __attribute__((ext_vector_type(8)))  float    v8f;

constexpr int N_ = 64, L_ = 2048, C_ = 256, M_ = 128;
constexpr int T_  = N_ * L_;      // 131072 tokens
constexpr int TT  = 64;           // tokens per tile
constexpr int GWG = 128;          // phase-A workgroups
constexpr int TPG = T_ / GWG;     // 1024 tokens per workgroup

__device__ inline v8f wzero() {
  v8f z;
#pragma unroll
  for (int i = 0; i < 8; ++i) z[i] = 0.f;
  return z;
}

__device__ inline v8f wmma_f16(v16h a, v16h b, v8f c) {
  // 8 args: (neg_a, A, neg_b, B, c_mod, C, reuse_a, reuse_b)
  return __builtin_amdgcn_wmma_f32_16x16x32_f16(false, a, false, b, (short)0, c,
                                                false, false);
}

// A operand: 16x32 f16 tile of row-major [rows x ld] at (row0, k0).
// lane l: row m = l&15, half h = l>>4; VGPR v holds K = 16h + 2v, +1
__device__ inline v16h load_a(const _Float16* p, int ld, int row0, int k0, int lane) {
  const int m = lane & 15, h = lane >> 4;
  v16h a;
#pragma unroll
  for (int v = 0; v < 8; ++v) {
    const int kb = k0 + h * 16 + v * 2;
    h2 pr = *(const h2*)(p + (row0 + m) * ld + kb);
    a[2 * v] = pr[0]; a[2 * v + 1] = pr[1];
  }
  return a;
}

// B operand (32x16) with B[k][n] = mat[n0+n][k0+k]  (transposed use; pairs contiguous)
// lane l: col n = l&15, half h = l>>4; VGPR v holds K = 16h + 2v, +1
__device__ inline v16h load_bt(const _Float16* p, int ld, int k0, int n0, int lane) {
  const int n = lane & 15, h = lane >> 4;
  v16h b;
#pragma unroll
  for (int v = 0; v < 8; ++v) {
    const int k = k0 + h * 16 + 2 * v;
    h2 pr = *(const h2*)(p + (n0 + n) * ld + k);
    b[2 * v] = pr[0]; b[2 * v + 1] = pr[1];
  }
  return b;
}

// B operand (32x16) with B[k][n] = mat[k0+k][n0+n]
__device__ inline v16h load_bn(const _Float16* p, int ld, int k0, int n0, int lane) {
  const int n = lane & 15, h = lane >> 4;
  v16h b;
#pragma unroll
  for (int v = 0; v < 8; ++v) {
    const int k = k0 + h * 16 + 2 * v;
    b[2 * v]     = p[(k    ) * ld + n0 + n];
    b[2 * v + 1] = p[(k + 1) * ld + n0 + n];
  }
  return b;
}

// C/D 16x16 f32 layout: lane l -> col n = l&15; VGPR v -> row v + 8*(l>>4)
__device__ inline void store_d(float* p, long ld, int row0, int col0, int lane, v8f d) {
  const int n = lane & 15, h = lane >> 4;
#pragma unroll
  for (int v = 0; v < 8; ++v)
    p[(long)(row0 + v + 8 * h) * ld + col0 + n] = d[v];
}

// ---------------- Phase A: attn_u = softmax(q @ mem^T); partial add_mem = attn_u^T @ q
__global__ __launch_bounds__(256) void k_update_partial(
    const float* __restrict__ q, const float* __restrict__ mem,
    float* __restrict__ partial) {
  __shared__ _Float16 sMem[M_ * C_];   // 64 KB
  __shared__ _Float16 sQ[TT * C_];     // 32 KB
  __shared__ float    sS[TT * M_];     // 32 KB
  __shared__ _Float16 sAT[M_ * TT];    // 16 KB  (attn^T tile, A operand)

  const int tid = threadIdx.x;
  const int wave = tid >> 5, lane = tid & 31;

  for (int i = tid; i < M_ * C_; i += 256) sMem[i] = (_Float16)mem[i];

  v8f accU[16];                        // wave owns M rows [wave*16, wave*16+16) x all C
#pragma unroll
  for (int j = 0; j < 16; ++j) accU[j] = wzero();

  const long t0wg = (long)blockIdx.x * TPG;
  for (int it = 0; it < TPG / TT; ++it) {
    const long t0 = t0wg + (long)it * TT;
    __syncthreads();                   // protect sQ/sAT from previous iteration
    for (int i = tid; i < TT * C_; i += 256) sQ[i] = (_Float16)q[t0 * C_ + i];
    __syncthreads();

    // logits: S(64x128) = q_tile(64xC) @ mem^T
    {
      const int rb  = wave >> 1;       // token row-block 0..3
      const int cb0 = (wave & 1) * 4;  // memory col-blocks cb0..cb0+3
      v8f s[4];
#pragma unroll
      for (int j = 0; j < 4; ++j) s[j] = wzero();
      for (int k0 = 0; k0 < C_; k0 += 32) {
        v16h a = load_a(sQ, C_, rb * 16, k0, lane);
#pragma unroll
        for (int j = 0; j < 4; ++j) {
          v16h b = load_bt(sMem, C_, k0, (cb0 + j) * 16, lane);
          s[j] = wmma_f16(a, b, s[j]);
        }
      }
#pragma unroll
      for (int j = 0; j < 4; ++j)
        store_d(sS, M_, rb * 16, (cb0 + j) * 16, lane, s[j]);
    }
    __syncthreads();

    // row softmax + eps clamp; store transposed f16 (rows = memory slots, K = tokens)
    if (tid < TT) {
      float mx = -3.4e38f;
      for (int k = 0; k < M_; ++k) mx = fmaxf(mx, sS[tid * M_ + k]);
      float sum = 0.f;
      for (int k = 0; k < M_; ++k) sum += __expf(sS[tid * M_ + k] - mx);
      const float inv = 1.f / sum;
      for (int k = 0; k < M_; ++k) {
        float pz = fmaxf(__expf(sS[tid * M_ + k] - mx) * inv, 1e-8f);
        sAT[k * TT + tid] = (_Float16)pz;
      }
    }
    __syncthreads();

    // add_mem partial (M x C) += attn^T(M x TT) @ q(TT x C)
    for (int k0 = 0; k0 < TT; k0 += 32) {
      v16h a = load_a(sAT, TT, wave * 16, k0, lane);
#pragma unroll
      for (int j = 0; j < 16; ++j) {
        v16h b = load_bn(sQ, C_, k0, j * 16, lane);
        accU[j] = wmma_f16(a, b, accU[j]);
      }
    }
  }

  float* pp = partial + (long)blockIdx.x * (M_ * C_);
#pragma unroll
  for (int j = 0; j < 16; ++j) store_d(pp, C_, wave * 16, j * 16, lane, accU[j]);
}

// ---------------- Phase A2: reduce partials, gate, momentum blend, row L2-normalize
__global__ __launch_bounds__(256) void k_finalize(
    const float* __restrict__ mem, const float* __restrict__ Uw,
    const float* __restrict__ Ub, const float* __restrict__ Ww,
    const float* __restrict__ Wb, const float* __restrict__ partial,
    float* __restrict__ newmem, _Float16* __restrict__ nmh) {
  __shared__ float sMrow[C_], sArow[C_], red[256];
  const int m = blockIdx.x, c = threadIdx.x;

  float am = 0.f;
  for (int g = 0; g < GWG; ++g) am += partial[(long)g * (M_ * C_) + m * C_ + c];
  const float mv = mem[m * C_ + c];
  sMrow[c] = mv; sArow[c] = am;
  __syncthreads();

  float a1 = 0.f, a2 = 0.f;
  for (int k = 0; k < C_; ++k) {
    a1 = fmaf(sMrow[k], Uw[c * C_ + k], a1);   // (mem @ U_w^T)[m][c]
    a2 = fmaf(sArow[k], Ww[c * C_ + k], a2);   // (add_mem @ W_w^T)[m][c]
  }
  const float gt = 1.f / (1.f + __expf(-(a1 + Ub[c] + a2 + Wb[c])));
  float nv = 0.9f * mv + 0.1f * ((1.f - gt) * mv + gt * am);

  red[c] = nv * nv;
  for (int s = 128; s > 0; s >>= 1) {
    __syncthreads();
    if (c < s) red[c] += red[c + s];
  }
  __syncthreads();
  const float nrm = fmaxf(sqrtf(red[0]), 1e-12f);
  nv /= nrm;
  newmem[m * C_ + c] = nv;
  nmh[m * C_ + c] = (_Float16)nv;
}

// ---------------- Phase B: read path (softmax -> hard-shrink-relu -> L1 norm -> GEMM)
__global__ __launch_bounds__(256) void k_read(
    const float* __restrict__ q, const _Float16* __restrict__ nmh,
    float* __restrict__ out, float* __restrict__ attn) {
  __shared__ _Float16 sNM[M_ * C_];  // 64 KB
  __shared__ _Float16 sQ[TT * C_];   // 32 KB
  __shared__ float    sS[TT * M_];   // 32 KB
  __shared__ _Float16 sA[TT * M_];   // 16 KB
  const int tid = threadIdx.x, wave = tid >> 5, lane = tid & 31;
  const long t0 = (long)blockIdx.x * TT;

  for (int i = tid; i < M_ * C_; i += 256) sNM[i] = nmh[i];
  for (int i = tid; i < TT * C_; i += 256) {
    const float v = q[t0 * C_ + i];
    sQ[i] = (_Float16)v;
    const int tt = i >> 8, cc = i & (C_ - 1);        // C_ == 256
    out[(t0 + tt) * (2 * C_) + cc] = v;              // out[:, :C] = q
  }
  __syncthreads();

  { // S = q_tile @ new_mem^T
    const int rb  = wave >> 1;
    const int cb0 = (wave & 1) * 4;
    v8f s[4];
#pragma unroll
    for (int j = 0; j < 4; ++j) s[j] = wzero();
    for (int k0 = 0; k0 < C_; k0 += 32) {
      v16h a = load_a(sQ, C_, rb * 16, k0, lane);
#pragma unroll
      for (int j = 0; j < 4; ++j) {
        v16h b = load_bt(sNM, C_, k0, (cb0 + j) * 16, lane);
        s[j] = wmma_f16(a, b, s[j]);
      }
    }
#pragma unroll
    for (int j = 0; j < 4; ++j)
      store_d(sS, M_, rb * 16, (cb0 + j) * 16, lane, s[j]);
  }
  __syncthreads();

  if (tid < TT) {
    float mx = -3.4e38f;
    for (int k = 0; k < M_; ++k) mx = fmaxf(mx, sS[tid * M_ + k]);
    float sum = 0.f;
    for (int k = 0; k < M_; ++k) sum += __expf(sS[tid * M_ + k] - mx);
    const float inv = 1.f / sum;
    float l1 = 0.f;
    for (int k = 0; k < M_; ++k) {
      const float a = __expf(sS[tid * M_ + k] - mx) * inv;
      const float d = a - 0.0025f;                        // hard_shrink_relu
      const float v = (d > 0.f ? d : 0.f) * a / (fabsf(d) + 1e-12f);
      sS[tid * M_ + k] = v;
      l1 += fabsf(v);
    }
    const float inv2 = 1.f / fmaxf(l1, 1e-12f);
    for (int k = 0; k < M_; ++k) {
      const float v = sS[tid * M_ + k] * inv2;
      attn[(t0 + tid) * M_ + k] = v;
      sA[tid * M_ + k] = (_Float16)v;
    }
  }
  __syncthreads();

  { // add_memory(64x256) = attn(64x128) @ new_mem(128x256) -> out[:, C:2C]
    const int rb  = wave & 3;
    const int cb0 = (wave >> 2) * 8;
    v8f s[8];
#pragma unroll
    for (int j = 0; j < 8; ++j) s[j] = wzero();
    for (int k0 = 0; k0 < M_; k0 += 32) {
      v16h a = load_a(sA, M_, rb * 16, k0, lane);
#pragma unroll
      for (int j = 0; j < 8; ++j) {
        v16h b = load_bn(sNM, C_, k0, (cb0 + j) * 16, lane);
        s[j] = wmma_f16(a, b, s[j]);
      }
    }
    const int n = lane & 15, h = lane >> 4;
#pragma unroll
    for (int j = 0; j < 8; ++j)
#pragma unroll
      for (int v = 0; v < 8; ++v)
        out[(t0 + rb * 16 + v + 8 * h) * (2 * C_) + C_ + (cb0 + j) * 16 + n] = s[j][v];
  }
}

extern "C" void kernel_launch(void* const* d_in, const int* in_sizes, int n_in,
                              void* d_out, int out_size, void* d_ws, size_t ws_size,
                              hipStream_t stream) {
  (void)in_sizes; (void)n_in; (void)out_size; (void)ws_size;
  const float* q   = (const float*)d_in[0];
  const float* mem = (const float*)d_in[1];
  const float* Uw  = (const float*)d_in[2];
  const float* Ub  = (const float*)d_in[3];
  const float* Ww  = (const float*)d_in[4];
  const float* Wb  = (const float*)d_in[5];

  float* out    = (float*)d_out;                  // T x 2C
  float* attn   = out + (long)T_ * 2 * C_;        // T x M
  float* newmem = attn + (long)T_ * M_;           // M x C

  float*    partial = (float*)d_ws;               // GWG x M x C  (~16.8 MB)
  _Float16* nmh     = (_Float16*)(partial + (long)GWG * M_ * C_);

  k_update_partial<<<GWG, 256, 0, stream>>>(q, mem, partial);
  k_finalize<<<M_, 256, 0, stream>>>(mem, Uw, Ub, Ww, Wb, partial, newmem, nmh);
  k_read<<<T_ / TT, 256, 0, stream>>>(q, nmh, out, attn);
}
